// OutputLayerTF_37778532336020
// MI455X (gfx1250) — matrix operand order repaired and checked by reading, not compiled, not run on hardware
//
#include <hip/hip_runtime.h>
#include <hip/hip_bf16.h>

// ---------------------------------------------------------------------------
// OutputLayerTF for MI455X (gfx1250, wave32, WMMA)
//
// B=4, N=2048, D=512, Y=8921.  ~300 GFLOP of fp32 GEMM, all data L2-resident
// (~70MB) -> purely matrix-issue bound.  Fast path: fp32-accurate GEMM via
// bf16x3 split on V_WMMA_F32_16X16X32_BF16 (8192 MAC/instr, 3 instrs per
// K=32 chunk) = ~2.7x fewer matrix issues than V_WMMA_F32_16X16X4_F32.
// Splits are precomputed (prep kernel -> d_ws) so the hot loops are pure
// b128 loads + WMMA.  Fallback: exact f32 WMMA path if ws is too small.
// ---------------------------------------------------------------------------

#define BB 4
#define NN 2048
#define DD 512
#define YY 8921
#define YTILES 558           // ceil(8921/16)
#define S_STRIDE 2052        // f32 score rows: stride%64==4 -> conflict-free
#define A_STRIDE 2072        // bf16 alpha rows (u16 units), 16B-friendly
#define U_STRIDE 520         // bf16 U rows (u16 units), 16B-friendly
#define BY (BB * YY)         // 35684
#define WS_NEED ((size_t)4 * BB * NN * DD * sizeof(unsigned short))

typedef __attribute__((ext_vector_type(2)))  float   v2f;
typedef __attribute__((ext_vector_type(8)))  float   v8f;
typedef __attribute__((ext_vector_type(8)))  __bf16  v8bf;
typedef __attribute__((ext_vector_type(16))) __bf16  v16bf;

static __device__ __forceinline__ v8f wmma_f32(v2f a, v2f b, v8f c) {
  return __builtin_amdgcn_wmma_f32_16x16x4_f32(
      false, a, false, b, (short)0, c, false, false);
}

static __device__ __forceinline__ v8f wmma_bf16(v16bf a, v16bf b, v8f c) {
  return __builtin_amdgcn_wmma_f32_16x16x32_bf16(
      false, a, false, b, (short)0, c, false, false);
}

// Two contiguous 8-element bf16 runs -> one 16-element fragment.
static __device__ __forceinline__ v16bf frag16(const unsigned short* p0,
                                               const unsigned short* p1) {
  v8bf a = *(const v8bf*)p0;
  v8bf b = *(const v8bf*)p1;
  return __builtin_shufflevector(a, b, 0, 1, 2, 3, 4, 5, 6, 7,
                                       8, 9, 10, 11, 12, 13, 14, 15);
}

// Truncation split: f == hi + lo to ~17 mantissa bits (exact hi).
static __device__ __forceinline__ void split_bf16(float f,
                                                  unsigned short& hi,
                                                  unsigned short& lo) {
  unsigned u  = __float_as_uint(f);
  unsigned hu = u & 0xffff0000u;
  hi = (unsigned short)(u >> 16);
  float rem = f - __uint_as_float(hu);
  lo = (unsigned short)(__float_as_uint(rem) >> 16);
}

// ---------------------------------------------------------------------------
// Prep: split x into bf16 hi/lo, row-major [B,N,D] and transposed [B,D,N].
// ---------------------------------------------------------------------------
__global__ __launch_bounds__(256) void prep_split_kernel(
    const float* __restrict__ x,
    unsigned short* __restrict__ xhi, unsigned short* __restrict__ xlo,
    unsigned short* __restrict__ xthi, unsigned short* __restrict__ xtlo) {
  __shared__ float tile[64][65];
  const int b  = blockIdx.z;
  const int n0 = blockIdx.y * 64;
  const int d0 = blockIdx.x * 64;
  const size_t boff = (size_t)b * NN * DD;

  for (int k = 0; k < 64 * 64; k += 256) {
    int idx = k + threadIdx.x;
    int nl = idx >> 6, dl = idx & 63;
    float v = x[boff + (size_t)(n0 + nl) * DD + d0 + dl];
    tile[nl][dl] = v;
    unsigned short hi, lo;
    split_bf16(v, hi, lo);
    size_t o = boff + (size_t)(n0 + nl) * DD + d0 + dl;
    xhi[o] = hi;
    xlo[o] = lo;
  }
  __syncthreads();
  for (int k = 0; k < 64 * 64; k += 256) {
    int idx = k + threadIdx.x;
    int dl = idx >> 6, nl = idx & 63;
    float v = tile[nl][dl];
    unsigned short hi, lo;
    split_bf16(v, hi, lo);
    size_t o = boff + (size_t)(d0 + dl) * NN + n0 + nl;
    xthi[o] = hi;
    xtlo[o] = lo;
  }
}

// ---------------------------------------------------------------------------
// Fast path: bf16x3 fused attention.
// ---------------------------------------------------------------------------
__global__ __launch_bounds__(256, 1) void attn_fused_bf16(
    const unsigned short* __restrict__ xhi,   // [B,N,D] bf16
    const unsigned short* __restrict__ xlo,
    const unsigned short* __restrict__ xthi,  // [B,D,N] bf16
    const unsigned short* __restrict__ xtlo,
    const float* __restrict__ y_flow,
    const float* __restrict__ U_w, const float* __restrict__ Ug_w,
    const float* __restrict__ final_w, const float* __restrict__ final_b,
    const float* __restrict__ gate_w, const float* __restrict__ gate_b,
    float* __restrict__ y_out) {
  __shared__ float Slds[16 * S_STRIDE];          // raw f32 scores
  __shared__ unsigned short Ahi[16 * A_STRIDE];  // alpha hi (unnormalized exp)
  __shared__ unsigned short Alo[16 * A_STRIDE];  // alpha lo
  __shared__ unsigned short Ubh[16 * U_STRIDE];  // U tile hi, [y][k]
  __shared__ unsigned short Ubl[16 * U_STRIDE];  // U tile lo
  __shared__ float inv_s[16];
  __shared__ float ypart[16];
  __shared__ float gpart[16];

  const int tid = threadIdx.x;
  const int wave = tid >> 5;
  const int lane = tid & 31;
  const int l = lane & 15;
  const int h = lane >> 4;
  const int ytile = blockIdx.x;
  const int b = blockIdx.y;
  const int ybase = ytile * 16;
  const size_t boff = (size_t)b * NN * DD;

  const unsigned short* xhib = xhi + boff;
  const unsigned short* xlob = xlo + boff;
  const unsigned short* xthib = xthi + boff;
  const unsigned short* xtlob = xtlo + boff;

  if (tid < 16) { ypart[tid] = 0.0f; gpart[tid] = 0.0f; }

  for (int br = 0; br < 2; ++br) {
    __syncthreads();

    // ---- load + split weight tile into [y][k] bf16 hi/lo ---------------
    const float* W = br ? Ug_w : U_w;
    for (int idx = tid; idx < 16 * DD; idx += 256) {
      int r = idx >> 9;
      int k = idx & (DD - 1);
      int yg = ybase + r; if (yg > YY - 1) yg = YY - 1;
      unsigned short hi, lo;
      split_bf16(W[(size_t)yg * DD + k], hi, lo);
      Ubh[r * U_STRIDE + k] = hi;
      Ubl[r * U_STRIDE + k] = lo;
    }
    __syncthreads();

    // ---- phase A: S^T = x * U^T via bf16x3, K = D = 512 ----------------
    for (int nb = wave * 16; nb < NN; nb += 128) {
      v8f c = {};
      const unsigned short* xh = xhib + (size_t)(nb + l) * DD;
      const unsigned short* xl = xlob + (size_t)(nb + l) * DD;
      const unsigned short* ubh = &Ubh[l * U_STRIDE];
      const unsigned short* ubl = &Ubl[l * U_STRIDE];
      for (int k0 = 0; k0 < DD; k0 += 32) {
        int ka = k0 + 8 * h;
        v16bf ah = frag16(xh + ka, xh + ka + 16);
        v16bf al = frag16(xl + ka, xl + ka + 16);
        int kb = k0 + 8 * h;
        v16bf bh = frag16(ubh + kb, ubh + kb + 16);
        v16bf bl = frag16(ubl + kb, ubl + kb + 16);
        c = wmma_bf16(ah, bh, c);
        c = wmma_bf16(ah, bl, c);
        c = wmma_bf16(al, bh, c);
      }
      #pragma unroll
      for (int i = 0; i < 8; ++i)
        Slds[l * S_STRIDE + nb + i + 8 * h] = c[i];
    }
    __syncthreads();

    // ---- phase B: softmax; write alpha = exp(s-max) as bf16 hi/lo ------
    for (int rr = 0; rr < 2; ++rr) {
      int r = wave * 2 + rr;
      float* row = &Slds[r * S_STRIDE];
      float mx = -3.402823466e38f;
      for (int n = lane; n < NN; n += 32) mx = fmaxf(mx, row[n]);
      for (int m = 16; m >= 1; m >>= 1) mx = fmaxf(mx, __shfl_xor(mx, m, 32));
      float sum = 0.0f;
      for (int n = lane; n < NN; n += 32) {
        float e = __expf(row[n] - mx);
        unsigned short hi, lo;
        split_bf16(e, hi, lo);
        Ahi[r * A_STRIDE + n] = hi;
        Alo[r * A_STRIDE + n] = lo;
        sum += e;
      }
      for (int m = 16; m >= 1; m >>= 1) sum += __shfl_xor(sum, m, 32);
      if (lane == 0) inv_s[r] = 1.0f / sum;
    }
    __syncthreads();

    // ---- phase C: m = alpha @ x via bf16x3, per-wave 64-wide d slice ---
    const int d0 = wave * 64;
    v8f acc[4] = {{}, {}, {}, {}};
    const unsigned short* arh = &Ahi[l * A_STRIDE];
    const unsigned short* arl = &Alo[l * A_STRIDE];
    for (int n0 = 0; n0 < NN; n0 += 32) {
      int na = n0 + 8 * h;
      v16bf ah = frag16(arh + na, arh + na + 16);
      v16bf al = frag16(arl + na, arl + na + 16);
      int nk = n0 + 8 * h;
      #pragma unroll
      for (int t = 0; t < 4; ++t) {
        int d = d0 + t * 16 + l;
        const unsigned short* pbh = xthib + (size_t)d * NN + nk;
        const unsigned short* pbl = xtlob + (size_t)d * NN + nk;
        v16bf bh = frag16(pbh, pbh + 16);
        v16bf bl = frag16(pbl, pbl + 16);
        acc[t] = wmma_bf16(ah, bh, acc[t]);
        acc[t] = wmma_bf16(ah, bl, acc[t]);
        acc[t] = wmma_bf16(al, bh, acc[t]);
      }
    }

    float invr[8];
    #pragma unroll
    for (int i = 0; i < 8; ++i) invr[i] = inv_s[i + 8 * h];

    float part[8] = {0, 0, 0, 0, 0, 0, 0, 0};
    #pragma unroll
    for (int t = 0; t < 4; ++t) {
      int d = d0 + t * 16 + l;
      float gw = gate_w[d];
      #pragma unroll
      for (int i = 0; i < 8; ++i) {
        float mval = acc[t][i] * invr[i];
        float w;
        if (br == 0) {
          int yg = ybase + i + 8 * h; if (yg > YY - 1) yg = YY - 1;
          w = final_w[(size_t)yg * DD + d];
        } else {
          w = gw;
        }
        part[i] += mval * w;
      }
    }
    #pragma unroll
    for (int i = 0; i < 8; ++i) {
      float p = part[i];
      p += __shfl_xor(p, 1, 32);
      p += __shfl_xor(p, 2, 32);
      p += __shfl_xor(p, 4, 32);
      p += __shfl_xor(p, 8, 32);
      if (l == 0) atomicAdd(br ? &gpart[i + 8 * h] : &ypart[i + 8 * h], p);
    }
  }

  __syncthreads();

  if (tid < 16) {
    int yg = ybase + tid;
    if (yg < YY) {
      float g = tanhf(gpart[tid] + gate_b[0]);
      y_out[(size_t)b * YY + yg] =
          ypart[tid] + final_b[yg] + y_flow[(size_t)b * YY + yg] * g;
    }
  }
}

// ---------------------------------------------------------------------------
// Fallback: exact fp32 WMMA path (no workspace needed).
// ---------------------------------------------------------------------------
__global__ __launch_bounds__(256, 1) void attn_fused_f32(
    const float* __restrict__ x,
    const float* __restrict__ y_flow,
    const float* __restrict__ U_w, const float* __restrict__ Ug_w,
    const float* __restrict__ final_w, const float* __restrict__ final_b,
    const float* __restrict__ gate_w, const float* __restrict__ gate_b,
    float* __restrict__ y_out) {
  __shared__ float Slds[16 * S_STRIDE];
  __shared__ float Ulds[DD * 16];
  __shared__ float inv_s[16];
  __shared__ float ypart[16];
  __shared__ float gpart[16];

  const int tid = threadIdx.x;
  const int wave = tid >> 5;
  const int lane = tid & 31;
  const int l = lane & 15;
  const int h = lane >> 4;
  const int ytile = blockIdx.x;
  const int b = blockIdx.y;
  const int ybase = ytile * 16;

  const float* xb = x + (size_t)b * NN * DD;

  if (tid < 16) { ypart[tid] = 0.0f; gpart[tid] = 0.0f; }

  for (int br = 0; br < 2; ++br) {
    __syncthreads();

    const float* W = br ? Ug_w : U_w;
    for (int idx = tid; idx < 16 * DD; idx += 256) {
      int r = idx >> 9;
      int k = idx & (DD - 1);
      int yg = ybase + r; if (yg > YY - 1) yg = YY - 1;
      Ulds[k * 16 + r] = W[(size_t)yg * DD + k];
    }
    __syncthreads();

    for (int nb = wave * 16; nb < NN; nb += 128) {
      v8f c = {};
      const float* xr = xb + (size_t)(nb + l) * DD + 2 * h;
      #pragma unroll 4
      for (int k0 = 0; k0 < DD; k0 += 4) {
        v2f a = *(const v2f*)(xr + k0);
        int kk = k0 + 2 * h;
        v2f bv = { Ulds[kk * 16 + l], Ulds[(kk + 1) * 16 + l] };
        c = wmma_f32(a, bv, c);
      }
      #pragma unroll
      for (int i = 0; i < 8; ++i)
        Slds[l * S_STRIDE + nb + i + 8 * h] = c[i];
    }
    __syncthreads();

    for (int rr = 0; rr < 2; ++rr) {
      int r = wave * 2 + rr;
      float* row = &Slds[r * S_STRIDE];
      float mx = -3.402823466e38f;
      for (int n = lane; n < NN; n += 32) mx = fmaxf(mx, row[n]);
      for (int m = 16; m >= 1; m >>= 1) mx = fmaxf(mx, __shfl_xor(mx, m, 32));
      float sum = 0.0f;
      for (int n = lane; n < NN; n += 32) {
        float e = __expf(row[n] - mx);
        row[n] = e;
        sum += e;
      }
      for (int m = 16; m >= 1; m >>= 1) sum += __shfl_xor(sum, m, 32);
      if (lane == 0) inv_s[r] = 1.0f / sum;
    }
    __syncthreads();

    const int d0 = wave * 64;
    v8f acc[4] = {{}, {}, {}, {}};
    const float* xcol = xb + d0 + l;
    for (int n0 = 0; n0 < NN; n0 += 4) {
      v2f a = *(const v2f*)(&Slds[l * S_STRIDE + n0 + 2 * h]);
      size_t base = (size_t)(n0 + 2 * h) * DD;
      #pragma unroll
      for (int t = 0; t < 4; ++t) {
        v2f bv = { xcol[base + t * 16], xcol[base + DD + t * 16] };
        acc[t] = wmma_f32(a, bv, acc[t]);
      }
    }

    float invr[8];
    #pragma unroll
    for (int i = 0; i < 8; ++i) invr[i] = inv_s[i + 8 * h];

    float part[8] = {0, 0, 0, 0, 0, 0, 0, 0};
    #pragma unroll
    for (int t = 0; t < 4; ++t) {
      int d = d0 + t * 16 + l;
      float gw = gate_w[d];
      #pragma unroll
      for (int i = 0; i < 8; ++i) {
        float mval = acc[t][i] * invr[i];
        float w;
        if (br == 0) {
          int yg = ybase + i + 8 * h; if (yg > YY - 1) yg = YY - 1;
          w = final_w[(size_t)yg * DD + d];
        } else {
          w = gw;
        }
        part[i] += mval * w;
      }
    }
    #pragma unroll
    for (int i = 0; i < 8; ++i) {
      float p = part[i];
      p += __shfl_xor(p, 1, 32);
      p += __shfl_xor(p, 2, 32);
      p += __shfl_xor(p, 4, 32);
      p += __shfl_xor(p, 8, 32);
      if (l == 0) atomicAdd(br ? &gpart[i + 8 * h] : &ypart[i + 8 * h], p);
    }
  }

  __syncthreads();

  if (tid < 16) {
    int yg = ybase + tid;
    if (yg < YY) {
      float g = tanhf(gpart[tid] + gate_b[0]);
      y_out[(size_t)b * YY + yg] =
          ypart[tid] + final_b[yg] + y_flow[(size_t)b * YY + yg] * g;
    }
  }
}

// ---------------------------------------------------------------------------
// Loss: mean over B*Y of logaddexp(0,y) - y*target   (stable form)
// ---------------------------------------------------------------------------
__global__ __launch_bounds__(256) void bce_loss_kernel(
    const float* __restrict__ yv, const float* __restrict__ target,
    float* __restrict__ loss_out) {
  const int tid = threadIdx.x;
  const int wave = tid >> 5;
  const int lane = tid & 31;
  float acc = 0.0f;
  for (int i = tid; i < BY; i += 256) {
    float v = yv[i];
    float t = target[i];
    acc += fmaxf(v, 0.0f) + log1pf(__expf(-fabsf(v))) - v * t;
  }
  for (int m = 16; m >= 1; m >>= 1) acc += __shfl_xor(acc, m, 32);
  __shared__ float wsum[8];
  if (lane == 0) wsum[wave] = acc;
  __syncthreads();
  if (tid == 0) {
    float s = 0.0f;
    #pragma unroll
    for (int w = 0; w < 8; ++w) s += wsum[w];
    *loss_out = s / (float)BY;
  }
}

extern "C" void kernel_launch(void* const* d_in, const int* in_sizes, int n_in,
                              void* d_out, int out_size, void* d_ws, size_t ws_size,
                              hipStream_t stream) {
  const float* x       = (const float*)d_in[0];
  const float* target  = (const float*)d_in[1];
  const float* y_flow  = (const float*)d_in[2];
  const float* U_w     = (const float*)d_in[3];
  const float* Ug_w    = (const float*)d_in[4];
  const float* final_w = (const float*)d_in[5];
  const float* final_b = (const float*)d_in[6];
  const float* gate_w  = (const float*)d_in[7];
  const float* gate_b  = (const float*)d_in[8];
  float* out = (float*)d_out;

  dim3 grid(YTILES, BB);

  if (d_ws != nullptr && ws_size >= WS_NEED) {
    const size_t elems = (size_t)BB * NN * DD;
    unsigned short* xhi  = (unsigned short*)d_ws;
    unsigned short* xlo  = xhi + elems;
    unsigned short* xthi = xlo + elems;
    unsigned short* xtlo = xthi + elems;

    prep_split_kernel<<<dim3(DD / 64, NN / 64, BB), 256, 0, stream>>>(
        x, xhi, xlo, xthi, xtlo);
    attn_fused_bf16<<<grid, 256, 0, stream>>>(
        xhi, xlo, xthi, xtlo, y_flow, U_w, Ug_w,
        final_w, final_b, gate_w, gate_b, out);
  } else {
    attn_fused_f32<<<grid, 256, 0, stream>>>(
        x, y_flow, U_w, Ug_w, final_w, final_b, gate_w, gate_b, out);
  }

  bce_loss_kernel<<<1, 256, 0, stream>>>(out, target, out + (size_t)BY);
}
